// MultiHeadAttention_62096637165666
// MI455X (gfx1250) — matrix operand order repaired and checked by reading, not compiled, not run on hardware
//
#include <hip/hip_runtime.h>

#define DEV __device__ __forceinline__

typedef __attribute__((ext_vector_type(16))) __bf16 v16bf;
typedef __attribute__((ext_vector_type(8)))  __bf16 v8bf;
typedef __attribute__((ext_vector_type(8)))  float  v8f;

// types for the async global->LDS builtin (exact param types per hipcc diag)
typedef int v4i_g __attribute__((vector_size(16)));
typedef __attribute__((address_space(1))) v4i_g as1_v4i;   // global int4
typedef __attribute__((address_space(3))) v4i_g as3_v4i;   // LDS int4

constexpr int B_  = 2;
constexpr int L_  = 2048;
constexpr int DM  = 1024;   // d_model
constexpr int H_  = 16;     // heads
constexpr int DQ  = 64;     // d_qkv
constexpr int BL  = B_ * L_;

DEV int lane_id() { return (int)(threadIdx.x & 31u); }

DEV v8f wmma_bf16(v16bf a, v16bf b, v8f c) {
  // D = A(16x32) * B(32x16) + C, f32 accumulate
  return __builtin_amdgcn_wmma_f32_16x16x32_bf16(false, a, false, b,
                                                 (short)0, c, false, false);
}

DEV v16bf frag_cat(v8bf lo, v8bf hi) {
  v16bf r;
#pragma unroll
  for (int i = 0; i < 8; ++i) { r[i] = lo[i]; r[i + 8] = hi[i]; }
  return r;
}

// A fragment: 16(M) x 32(K) tile from row-major matrix, row stride ld (elements).
DEV v16bf load_frag_a(const __bf16* base, int ld) {
  int lane = lane_id();
  const __bf16* p = base + (lane & 15) * ld + ((lane < 16) ? 0 : 8);
  return frag_cat(*(const v8bf*)p, *(const v8bf*)(p + 16));
}

// B fragment: 32(K) x 16(N), storage is TRANSPOSED [N][K] row-major, stride ld.
DEV v16bf load_frag_b(const __bf16* base, int ld) {
  int lane = lane_id();
  const __bf16* p = base + (lane & 15) * ld + ((lane < 16) ? 0 : 8);
  return frag_cat(*(const v8bf*)p, *(const v8bf*)(p + 16));
}

// C/D layout: elem r -> (M = r + (lane<16?0:8), N = lane&15)
DEV void store_c_bf16(__bf16* base, int ld, v8f c) {
  int lane = lane_id();
  int n = lane & 15, mo = (lane < 16) ? 0 : 8;
#pragma unroll
  for (int r = 0; r < 8; ++r) base[(mo + r) * ld + n] = (__bf16)c[r];
}

// store transposed: element (M,N) -> base[N*ld + M]; 8 contiguous M per lane.
DEV void store_c_bf16_T(__bf16* base, int ld, v8f c) {
  int lane = lane_id();
  int n = lane & 15, mo = (lane < 16) ? 0 : 8;
  v8bf v;
#pragma unroll
  for (int r = 0; r < 8; ++r) v[r] = (__bf16)c[r];
  *(v8bf*)(base + n * ld + mo) = v;
}

// ---- CDNA5 async global->LDS copy (ASYNCcnt-tracked DMA path) ----------------
DEV void async_load_b128(const __bf16* g, __bf16* l) {
#if __has_builtin(__builtin_amdgcn_global_load_async_to_lds_b128)
  __builtin_amdgcn_global_load_async_to_lds_b128(
      (as1_v4i*)g, (as3_v4i*)l, 0, 0);
#else
  unsigned lofs = (unsigned)(__UINTPTR_TYPE__)(as3_v4i*)l;
  asm volatile("global_load_async_to_lds_b128 %0, %1, off"
               :: "v"(lofs), "v"(g) : "memory");
#endif
}

template <int N> DEV void wait_asynccnt() {
#if __has_builtin(__builtin_amdgcn_s_wait_asynccnt)
  __builtin_amdgcn_s_wait_asynccnt(N);
#else
  asm volatile("s_wait_asynccnt %0" :: "i"(N) : "memory");
#endif
}

// ---------------- stage 0: precision conversion / weight transposes ----------

__global__ void k_cvt_bf16(const float* __restrict__ x, __bf16* __restrict__ y, int n) {
  int i = blockIdx.x * blockDim.x + threadIdx.x;
  if (i < n) y[i] = (__bf16)x[i];
}

// w[H][DM][DQ] -> wt[H][DQ][DM]   (so contraction dim DM is contiguous)
__global__ void k_transpose_w(const float* __restrict__ w, __bf16* __restrict__ wt) {
  int i = blockIdx.x * blockDim.x + threadIdx.x;          // over H*DQ*DM
  if (i >= H_ * DQ * DM) return;
  int c = i & (DM - 1);
  int d = (i >> 10) & (DQ - 1);
  int h = i >> 16;
  wt[i] = (__bf16)w[(h * DM + c) * DQ + d];
}

// w_o[H][DQ][DM] -> wt[DM][H*DQ]  (flat contraction h*d contiguous)
__global__ void k_transpose_wo(const float* __restrict__ w, __bf16* __restrict__ wt) {
  int i = blockIdx.x * blockDim.x + threadIdx.x;          // over DM*DM
  if (i >= DM * DM) return;
  int hd = i & (DM - 1);
  int m  = i >> 10;
  wt[i] = (__bf16)w[hd * DM + m];
}

// ---------------- stage 1: fused Q/K/V projection -----------------------------
__global__ __launch_bounds__(256) void k_qkv(
    const __bf16* __restrict__ Xb, const __bf16* __restrict__ WqT,
    const __bf16* __restrict__ WkT, const __bf16* __restrict__ WvT,
    __bf16* __restrict__ Qb, __bf16* __restrict__ Kb, __bf16* __restrict__ Vt) {
  int wid = blockIdx.x * (blockDim.x >> 5) + (threadIdx.x >> 5);
  int lt = wid & 127;            // L/16 tiles
  int h  = (wid >> 7) & (H_ - 1);
  int b  = wid >> 11;
  const __bf16* xrow = Xb + (size_t)(b * L_ + lt * 16) * DM;
  const __bf16* wq = WqT + (size_t)h * DQ * DM;
  const __bf16* wk = WkT + (size_t)h * DQ * DM;
  const __bf16* wv = WvT + (size_t)h * DQ * DM;

  v8f cq[4] = {}, ck[4] = {}, cv[4] = {};
  for (int k0 = 0; k0 < DM; k0 += 32) {
    v16bf a = load_frag_a(xrow + k0, DM);
#pragma unroll
    for (int nf = 0; nf < 4; ++nf) {
      cq[nf] = wmma_bf16(a, load_frag_b(wq + nf * 16 * DM + k0, DM), cq[nf]);
      ck[nf] = wmma_bf16(a, load_frag_b(wk + nf * 16 * DM + k0, DM), ck[nf]);
      cv[nf] = wmma_bf16(a, load_frag_b(wv + nf * 16 * DM + k0, DM), cv[nf]);
    }
  }
  int bh = b * H_ + h;
  __bf16* qb = Qb + (size_t)(bh * L_ + lt * 16) * DQ;   // [bh][l][d]
  __bf16* kb = Kb + (size_t)(bh * L_ + lt * 16) * DQ;   // [bh][key][d]
  __bf16* vt = Vt + (size_t)bh * DQ * L_ + lt * 16;     // [bh][d][l]
#pragma unroll
  for (int nf = 0; nf < 4; ++nf) {
    store_c_bf16(qb + nf * 16, DQ, cq[nf]);
    store_c_bf16(kb + nf * 16, DQ, ck[nf]);
    store_c_bf16_T(vt + (size_t)nf * 16 * L_, L_, cv[nf]);
  }
}

// ---------------- stage 2: flash attention with async-LDS K/V staging --------
// Block = 8 waves sharing (b,h); each wave owns 16 queries (128 q / block).
// K/V tiles (32 keys) are DMA'd to LDS once per block (double-buffered) and
// consumed by all 8 waves; DMA of tile i+1 overlaps compute of tile i.
__global__ __launch_bounds__(256) void k_attn(
    const __bf16* __restrict__ Qb, const __bf16* __restrict__ Kb,
    const __bf16* __restrict__ Vt, const unsigned char* __restrict__ mask,
    __bf16* __restrict__ Ob) {
  __shared__ __bf16 kbuf[2][32 * DQ];   // [key][d]   4KB each
  __shared__ __bf16 vbuf[2][DQ * 32];   // [d][key]   4KB each
  __shared__ __bf16 pbuf[8][16 * 32];   // P staging per wave

  int warp = threadIdx.x >> 5;
  int wid0 = blockIdx.x * 8;
  int qt = (wid0 & 127) + warp;
  int h  = (wid0 >> 7) & (H_ - 1);
  int b  = wid0 >> 11;
  int bh = b * H_ + h;
  int lane = lane_id();
  int n = lane & 15, mo = (lane < 16) ? 0 : 8;
  int t = threadIdx.x;

  const __bf16* kmat = Kb + (size_t)bh * L_ * DQ;
  const __bf16* vmat = Vt + (size_t)bh * DQ * L_;
  const unsigned char* mk = mask + b * L_;

  const __bf16* qbase = Qb + (size_t)(bh * L_ + qt * 16) * DQ;
  v16bf aq0 = load_frag_a(qbase, DQ);        // d 0..31
  v16bf aq1 = load_frag_a(qbase + 32, DQ);   // d 32..63

  v8f o0 = {}, o1 = {}, o2 = {}, o3 = {};
  float mrow[8], lrow[8];
#pragma unroll
  for (int r = 0; r < 8; ++r) { mrow[r] = -3.0e38f; lrow[r] = 0.0f; }

  const float scale = 0.125f;  // 1/sqrt(64)
  constexpr int NCH = L_ / 32;

  // cooperative prefetch of one 32-key chunk into buffer p
  auto prefetch = [&](int p, int kb) {
    // K chunk: 32x64 contiguous bf16 (4KB): lane t copies 16B
    async_load_b128(kmat + (size_t)kb * DQ + t * 8, &kbuf[p][t * 8]);
    // V chunk: 64 rows (d) x 32 keys from Vt[d][L]: 4 x 16B per row
    int d = t >> 2, part = t & 3;
    async_load_b128(vmat + (size_t)d * L_ + kb + part * 8,
                    &vbuf[p][d * 32 + part * 8]);
  };

  prefetch(0, 0);
  for (int i = 0; i < NCH; ++i) {
    int cur = i & 1;
    int kb = i * 32;
    if (i + 1 < NCH) {
      prefetch(cur ^ 1, kb + 32);
      wait_asynccnt<2>();      // own 2 loads of chunk i complete (in-order)
    } else {
      wait_asynccnt<0>();
    }
    __syncthreads();           // everyone's chunk-i data visible in LDS

    const __bf16* Kl = &kbuf[cur][0];
    const __bf16* Vl = &vbuf[cur][0];

    // S = Q * K^T (two 16x16 C-frags over 32 keys), contraction d=64
    v8f s0 = {}, s1 = {};
    s0 = wmma_bf16(aq0, load_frag_b(Kl, DQ), s0);
    s0 = wmma_bf16(aq1, load_frag_b(Kl + 32, DQ), s0);
    s1 = wmma_bf16(aq0, load_frag_b(Kl + 16 * DQ, DQ), s1);
    s1 = wmma_bf16(aq1, load_frag_b(Kl + 16 * DQ + 32, DQ), s1);

    float bias0 = mk[kb + n] ? 0.0f : -1.0e9f;       // key col = lane&15
    float bias1 = mk[kb + 16 + n] ? 0.0f : -1.0e9f;
#pragma unroll
    for (int r = 0; r < 8; ++r) {
      s0[r] = s0[r] * scale + bias0;
      s1[r] = s1[r] * scale + bias1;
      float v = fmaxf(s0[r], s1[r]);
      v = fmaxf(v, __shfl_xor(v, 1, 32));
      v = fmaxf(v, __shfl_xor(v, 2, 32));
      v = fmaxf(v, __shfl_xor(v, 4, 32));
      v = fmaxf(v, __shfl_xor(v, 8, 32));
      float mnew = fmaxf(mrow[r], v);
      float alpha = __expf(mrow[r] - mnew);
      mrow[r] = mnew;
      s0[r] = __expf(s0[r] - mnew);
      s1[r] = __expf(s1[r] - mnew);
      float rs = s0[r] + s1[r];
      rs += __shfl_xor(rs, 1, 32);
      rs += __shfl_xor(rs, 2, 32);
      rs += __shfl_xor(rs, 4, 32);
      rs += __shfl_xor(rs, 8, 32);
      lrow[r] = lrow[r] * alpha + rs;
      o0[r] *= alpha; o1[r] *= alpha; o2[r] *= alpha; o3[r] *= alpha;
      // stage P tile (16 rows x 32 keys, row-major) for C->A layout change
      pbuf[warp][(mo + r) * 32 + n]      = (__bf16)s0[r];
      pbuf[warp][(mo + r) * 32 + 16 + n] = (__bf16)s1[r];
    }
    asm volatile("s_wait_dscnt 0" ::: "memory");
    const __bf16* pb = &pbuf[warp][(lane & 15) * 32 + ((lane < 16) ? 0 : 8)];
    v16bf ap = frag_cat(*(const v8bf*)pb, *(const v8bf*)(pb + 16));
    // O += P * V  (V tile in LDS as [d][key])
    o0 = wmma_bf16(ap, load_frag_b(Vl, 32), o0);
    o1 = wmma_bf16(ap, load_frag_b(Vl + 16 * 32, 32), o1);
    o2 = wmma_bf16(ap, load_frag_b(Vl + 32 * 32, 32), o2);
    o3 = wmma_bf16(ap, load_frag_b(Vl + 48 * 32, 32), o3);

    __syncthreads();           // all waves done with buf[cur] before reuse
  }

  // normalize and store as [b][l][h][d] so stage 3 is one flat GEMM
  __bf16* ob = Ob + ((size_t)(b * L_ + qt * 16) * H_ + h) * DQ;
#pragma unroll
  for (int r = 0; r < 8; ++r) {
    float inv = 1.0f / lrow[r];
    size_t rowoff = (size_t)(mo + r) * (H_ * DQ);
    ob[rowoff + n]      = (__bf16)(o0[r] * inv);
    ob[rowoff + 16 + n] = (__bf16)(o1[r] * inv);
    ob[rowoff + 32 + n] = (__bf16)(o2[r] * inv);
    ob[rowoff + 48 + n] = (__bf16)(o3[r] * inv);
  }
}

// ---------------- stage 3: output projection (head-sum folded into GEMM) -----
__global__ __launch_bounds__(256) void k_oproj(
    const __bf16* __restrict__ Ob, const __bf16* __restrict__ WoT,
    float* __restrict__ res) {
  int wid = blockIdx.x * 8 + (threadIdx.x >> 5);
  int nt = wid & 15;          // 16 tiles of 64 cols
  int mt = wid >> 4;          // 256 tiles of 16 rows
  const __bf16* arow = Ob + (size_t)mt * 16 * DM;
  const __bf16* wb   = WoT + (size_t)nt * 64 * DM;
  v8f c[4] = {};
  for (int k0 = 0; k0 < DM; k0 += 32) {
    v16bf a = load_frag_a(arow + k0, DM);
#pragma unroll
    for (int nf = 0; nf < 4; ++nf)
      c[nf] = wmma_bf16(a, load_frag_b(wb + (size_t)nf * 16 * DM + k0, DM), c[nf]);
  }
  int lane = lane_id();
  int n = lane & 15, mo = (lane < 16) ? 0 : 8;
  float* rb = res + (size_t)mt * 16 * DM + nt * 64;
#pragma unroll
  for (int nf = 0; nf < 4; ++nf)
#pragma unroll
    for (int r = 0; r < 8; ++r)
      rb[(size_t)(mo + r) * DM + nf * 16 + n] = c[nf][r];
}

// ---------------- stage 4: residual add + LayerNorm --------------------------
__global__ __launch_bounds__(256) void k_ln(
    const float* __restrict__ res, const float* __restrict__ x,
    const float* __restrict__ gamma, const float* __restrict__ beta,
    float* __restrict__ out) {
  __shared__ float red[256];
  int row = blockIdx.x, t = threadIdx.x;
  const float* rr = res + (size_t)row * DM;
  const float* xr = x + (size_t)row * DM;
  float hv[4];
  float s = 0.0f;
#pragma unroll
  for (int i = 0; i < 4; ++i) { hv[i] = rr[t + i * 256] + xr[t + i * 256]; s += hv[i]; }
  red[t] = s; __syncthreads();
  for (int off = 128; off > 0; off >>= 1) {
    if (t < off) red[t] += red[t + off];
    __syncthreads();
  }
  float mu = red[0] * (1.0f / DM);
  __syncthreads();
  float v = 0.0f;
#pragma unroll
  for (int i = 0; i < 4; ++i) { float d = hv[i] - mu; v += d * d; }
  red[t] = v; __syncthreads();
  for (int off = 128; off > 0; off >>= 1) {
    if (t < off) red[t] += red[t + off];
    __syncthreads();
  }
  float rstd = rsqrtf(red[0] * (1.0f / DM) + 1e-5f);
  float* orow = out + (size_t)row * DM;
#pragma unroll
  for (int i = 0; i < 4; ++i) {
    int c = t + i * 256;
    orow[c] = (hv[i] - mu) * rstd * gamma[c] + beta[c];
  }
}

// ---------------- launcher ----------------------------------------------------
extern "C" void kernel_launch(void* const* d_in, const int* in_sizes, int n_in,
                              void* d_out, int out_size, void* d_ws, size_t ws_size,
                              hipStream_t stream) {
  (void)in_sizes; (void)n_in; (void)out_size; (void)ws_size;
  const float* x   = (const float*)d_in[0];
  const unsigned char* mask = (const unsigned char*)d_in[1];
  const float* w_q = (const float*)d_in[2];
  const float* w_k = (const float*)d_in[3];
  const float* w_v = (const float*)d_in[4];
  const float* w_o = (const float*)d_in[5];
  const float* g   = (const float*)d_in[6];
  const float* bta = (const float*)d_in[7];
  float* out = (float*)d_out;

  char* ws = (char*)d_ws;
  size_t off = 0;
  __bf16* Xb  = (__bf16*)(ws + off); off += (size_t)BL * DM * 2;        // 8 MB
  __bf16* WqT = (__bf16*)(ws + off); off += (size_t)H_ * DQ * DM * 2;   // 2 MB
  __bf16* WkT = (__bf16*)(ws + off); off += (size_t)H_ * DQ * DM * 2;
  __bf16* WvT = (__bf16*)(ws + off); off += (size_t)H_ * DQ * DM * 2;
  __bf16* WoT = (__bf16*)(ws + off); off += (size_t)DM * DM * 2;        // 2 MB
  __bf16* Qb  = (__bf16*)(ws + off); off += (size_t)BL * DM * 2;        // 8 MB
  __bf16* Kb  = (__bf16*)(ws + off); off += (size_t)BL * DM * 2;
  __bf16* Vt  = (__bf16*)(ws + off); off += (size_t)BL * DM * 2;
  __bf16* Ob  = (__bf16*)(ws + off); off += (size_t)BL * DM * 2;
  float*  res = (float*)(ws + off);  off += (size_t)BL * DM * 4;        // 16 MB

  k_cvt_bf16<<<(BL * DM) / 256, 256, 0, stream>>>(x, Xb, BL * DM);
  k_transpose_w<<<(H_ * DQ * DM) / 256, 256, 0, stream>>>(w_q, WqT);
  k_transpose_w<<<(H_ * DQ * DM) / 256, 256, 0, stream>>>(w_k, WkT);
  k_transpose_w<<<(H_ * DQ * DM) / 256, 256, 0, stream>>>(w_v, WvT);
  k_transpose_wo<<<(DM * DM) / 256, 256, 0, stream>>>(w_o, WoT);
  k_qkv<<<512, 256, 0, stream>>>(Xb, WqT, WkT, WvT, Qb, Kb, Vt);
  k_attn<<<512, 256, 0, stream>>>(Qb, Kb, Vt, mask, Ob);
  k_oproj<<<512, 256, 0, stream>>>(Ob, WoT, res);
  k_ln<<<BL, 256, 0, stream>>>(res, x, g, bta, out);
}
